// PageRank2_47047071760689
// MI455X (gfx1250) — matrix operand order repaired and checked by reading, not compiled, not run on hardware
//
#include <hip/hip_runtime.h>
#include <hip/hip_bf16.h>

// ---------------------------------------------------------------------------
// Problem constants (match reference)
// ---------------------------------------------------------------------------
#define NN   8192   // nodes
#define HIDD 256    // hidden dim
#define OUTD 64     // output dim
#define TOPK 500

typedef __attribute__((ext_vector_type(16))) __bf16 v16bf;
typedef __attribute__((ext_vector_type(8)))  float  v8f;
typedef __attribute__((ext_vector_type(4)))  int    v4i;

// Async global->LDS copy path (CDNA5 GLOBAL_LOAD_ASYNC_TO_LDS_*, ASYNCcnt).
// Probe round 2 revealed the prototype: (v4i AS1*, v4i AS3*, imm offset, imm cpol).
#if defined(__AMDGCN__) && \
    __has_builtin(__builtin_amdgcn_global_load_async_to_lds_b128) && \
    __has_builtin(__builtin_amdgcn_s_wait_asynccnt)
#define HAS_ASYNC_LDS 1
typedef __attribute__((address_space(1))) v4i gv4i;   // global v4i
typedef __attribute__((address_space(3))) v4i lv4i;   // LDS v4i
#else
#define HAS_ASYNC_LDS 0
#endif

// ---------------------------------------------------------------------------
// Fragment builders (CDNA5 WMMA bf16 16x16x32 layouts, wave32)
//
// A (16x32, MxK): lane L (L<16) holds row M=L, K = {k0..k0+7, k0+16..k0+23};
//                 lane L (>=16) holds row M=L-16, K = {k0+8..k0+15, k0+24..k0+31}.
// B (32x16, KxN): lane L (L<16) holds col N=L, K = {k0..k0+7, k0+16..k0+23};
//                 lane L (>=16) holds col N=L-16, K = {k0+8..k0+15, k0+24..k0+31}.
// C/D (16x16 f32): VGPR v = row m0+v (lanes 0-15) / m0+8+v (lanes 16-31),
//                  lane&15 = column.
// ---------------------------------------------------------------------------

__device__ __forceinline__ v16bf a_frag_f32(const float* __restrict__ A,
                                            int row, int ld, int k0, int lane) {
  const int s1 = k0 + ((lane & 16) ? 8 : 0);
  const float* p = A + (size_t)row * (size_t)ld;
  float buf[16];
  *(float4*)(&buf[0])  = *(const float4*)(p + s1);
  *(float4*)(&buf[4])  = *(const float4*)(p + s1 + 4);
  *(float4*)(&buf[8])  = *(const float4*)(p + s1 + 16);
  *(float4*)(&buf[12]) = *(const float4*)(p + s1 + 20);
  v16bf f;
#pragma unroll
  for (int e = 0; e < 16; ++e) f[e] = (__bf16)buf[e];
  return f;
}

__device__ __forceinline__ v16bf b_frag_f32_global(const float* __restrict__ W,
                                                   int n, int ldn, int k0, int lane) {
  const int kb = k0 + ((lane & 16) ? 8 : 0);
  v16bf f;
#pragma unroll
  for (int e = 0; e < 16; ++e) f[e] = (__bf16)W[(size_t)(kb + (e & 7) + ((e & 8) << 1)) * (size_t)ldn + n];
  return f;
}

// ---------------------------------------------------------------------------
// Generic small GEMM: C[M,N] = act(A[M,K] @ W[K,N] (+ bias))
// One wave per 16x16 tile, 8 waves per block. M,N,K multiples of 16/32.
// ---------------------------------------------------------------------------
__global__ void gemm_bias_act(const float* __restrict__ A,
                              const float* __restrict__ W,
                              const float* __restrict__ bias,
                              float* __restrict__ C,
                              int M, int K, int N, int do_bias, int do_relu) {
  const int wave = threadIdx.x >> 5;
  const int lane = threadIdx.x & 31;
  const int tilesN = N >> 4;
  const int tile = blockIdx.x * 8 + wave;
  const int m0 = (tile / tilesN) << 4;
  const int n0 = (tile % tilesN) << 4;
  const int n  = n0 + (lane & 15);
  const int arow = m0 + (lane & 15);

  v8f acc = {};
  for (int k0 = 0; k0 < K; k0 += 32) {
    v16bf af = a_frag_f32(A, arow, K, k0, lane);
    v16bf bf = b_frag_f32_global(W, n, N, k0, lane);
    acc = __builtin_amdgcn_wmma_f32_16x16x32_bf16(
        false, af, false, bf, (short)0, acc, false, false);
  }

  const int hi = (lane >> 4) & 1;
  float bb = do_bias ? bias[n] : 0.0f;
#pragma unroll
  for (int v = 0; v < 8; ++v) {
    int r = m0 + v + hi * 8;
    float x = acc[v] + bb;
    if (do_relu) x = x > 0.0f ? x : 0.0f;
    C[(size_t)r * (size_t)N + n] = x;
  }
}

// ---------------------------------------------------------------------------
// Per-row top-k threshold via radix histogram on float bits [30:20].
// filt = relu(adj*sim) >= 0, so bit pattern is monotone in value.
// thr[row] = lower bound of the bucket containing the 500th-largest value
// (0 if the row has < 500 positive entries -> keep everything positive).
// One 256-thread block per row.
// ---------------------------------------------------------------------------
__global__ void row_topk_thresh(const float* __restrict__ adj,
                                const float* __restrict__ sim,
                                float* __restrict__ thr) {
  __shared__ unsigned hist[2048];
  const int row = blockIdx.x;
  for (int i = threadIdx.x; i < 2048; i += 256) hist[i] = 0u;
  __syncthreads();

  const float* pa = adj + (size_t)row * NN;
  const float* ps = sim + (size_t)row * NN;
  for (int j = threadIdx.x; j < NN; j += 256) {
    float v = pa[j] * ps[j];
    if (v > 0.0f) {
      unsigned b = __float_as_uint(v) >> 20;   // sign==0 -> b < 2048
      atomicAdd(&hist[b], 1u);
    }
  }
  __syncthreads();

  if (threadIdx.x == 0) {
    unsigned cum = 0;
    float t = 0.0f;
    for (int b = 2047; b >= 0; --b) {
      cum += hist[b];
      if (cum >= TOPK) { t = __uint_as_float(((unsigned)b) << 20); break; }
    }
    thr[row] = t;
  }
}

// ---------------------------------------------------------------------------
// Fused top-k-masked GCN GEMM:
//   C[8192,256] = relu( mask(relu(adj*sim), thr) @ Y + b_gcn )
//
// new_conn is never materialized: A tiles are regenerated on the fly from
// adj/sim/thr (saves 512MB of HBM round-trip). Each wave computes a 16x64
// output strip (4 accumulators) so the expensive masked A fragment is reused
// 4x per WMMA group. 8 waves stack along M -> 128x64 block tile; the shared
// Y k-slab (32x64 f32, 8KB) is staged through LDS via CDNA5 async
// global->LDS copies (ASYNCcnt) when the toolchain exposes them.
// ---------------------------------------------------------------------------
__global__ void gcn_gemm_fused(const float* __restrict__ adj,
                               const float* __restrict__ sim,
                               const float* __restrict__ thr,
                               const float* __restrict__ Y,
                               const float* __restrict__ bgcn,
                               float* __restrict__ C) {
  __shared__ float Ysl[32 * 64];

  const int wave = threadIdx.x >> 5;
  const int lane = threadIdx.x & 31;
  const int m0 = blockIdx.x * 128 + wave * 16;
  const int n0 = blockIdx.y * 64;
  const int row = m0 + (lane & 15);
  const float tr = thr[row];

  const float* pa = adj + (size_t)row * NN;
  const float* ps = sim + (size_t)row * NN;

  // Staging coordinates: 2048 floats per slab, 8 per thread (2 x b128).
  const int si = threadIdx.x * 8;
  const int skk = si >> 6;        // k within slab (0..31)
  const int snn = si & 63;        // n within slab

  v8f acc[4];
#pragma unroll
  for (int s = 0; s < 4; ++s) acc[s] = (v8f){};

  for (int k0 = 0; k0 < NN; k0 += 32) {
    // ---- stage Y[k0:k0+32, n0:n0+64] into LDS --------------------------
    {
      const float* gsrc = &Y[(size_t)(k0 + skk) * HIDD + n0 + snn];
      float* ldst = &Ysl[si];
#if HAS_ASYNC_LDS
      __builtin_amdgcn_global_load_async_to_lds_b128(
          (gv4i*)gsrc, (lv4i*)ldst, 0, 0);
      __builtin_amdgcn_global_load_async_to_lds_b128(
          (gv4i*)gsrc, (lv4i*)ldst, 16, 0);
      __builtin_amdgcn_s_wait_asynccnt(0);
#else
      *(float4*)(ldst)     = *(const float4*)(gsrc);
      *(float4*)(ldst + 4) = *(const float4*)(gsrc + 4);
#endif
    }
    __syncthreads();

    // ---- regenerate masked new_conn A fragment -------------------------
    const int s1 = k0 + ((lane & 16) ? 8 : 0);
    float av[16], sv[16];
    *(float4*)(&av[0])  = *(const float4*)(pa + s1);
    *(float4*)(&av[4])  = *(const float4*)(pa + s1 + 4);
    *(float4*)(&av[8])  = *(const float4*)(pa + s1 + 16);
    *(float4*)(&av[12]) = *(const float4*)(pa + s1 + 20);
    *(float4*)(&sv[0])  = *(const float4*)(ps + s1);
    *(float4*)(&sv[4])  = *(const float4*)(ps + s1 + 4);
    *(float4*)(&sv[8])  = *(const float4*)(ps + s1 + 16);
    *(float4*)(&sv[12]) = *(const float4*)(ps + s1 + 20);
    if (k0 + 64 <= NN) {                       // speculative prefetch, next k-block
      __builtin_prefetch(pa + s1 + 32, 0, 3);
      __builtin_prefetch(ps + s1 + 32, 0, 3);
    }
    v16bf af;
#pragma unroll
    for (int e = 0; e < 16; ++e) {
      float x = av[e] * sv[e];
      x = x > 0.0f ? x : 0.0f;   // relu(adj*sim)
      x = x >= tr ? x : 0.0f;    // per-row top-k mask
      af[e] = (__bf16)x;
    }

    // ---- 4 WMMAs: reuse A fragment across the 16x64 N strip ------------
    const int kb = (lane & 16) ? 8 : 0;
    const int nl = lane & 15;
#pragma unroll
    for (int s = 0; s < 4; ++s) {
      v16bf bf;
#pragma unroll
      for (int e = 0; e < 16; ++e)
        bf[e] = (__bf16)Ysl[(kb + (e & 7) + ((e & 8) << 1)) * 64 + s * 16 + nl];
      acc[s] = __builtin_amdgcn_wmma_f32_16x16x32_bf16(
          false, af, false, bf, (short)0, acc[s], false, false);
    }
    __syncthreads();
  }

  // ---- epilogue: bias + relu + store 16x64 strip ------------------------
  const int hi = (lane >> 4) & 1;
#pragma unroll
  for (int s = 0; s < 4; ++s) {
    const int n = n0 + s * 16 + (lane & 15);
    const float bb = bgcn[n];
#pragma unroll
    for (int v = 0; v < 8; ++v) {
      int r = m0 + v + hi * 8;
      float x = acc[s][v] + bb;
      x = x > 0.0f ? x : 0.0f;
      C[(size_t)r * HIDD + n] = x;
    }
  }
}

// ---------------------------------------------------------------------------
// Launch: hp = relu(h@Wp+bp); Y = hp@Wg; thr = rowtopk(adj,sim);
//         fg = relu(mask@Y + bg); out = relu(fg@Wdt + bdt)
// ---------------------------------------------------------------------------
extern "C" void kernel_launch(void* const* d_in, const int* in_sizes, int n_in,
                              void* d_out, int out_size, void* d_ws, size_t ws_size,
                              hipStream_t stream) {
  (void)in_sizes; (void)n_in; (void)out_size; (void)ws_size;

  const float* h      = (const float*)d_in[0];
  const float* adj    = (const float*)d_in[1];
  const float* simlar = (const float*)d_in[2];
  const float* W_proj = (const float*)d_in[3];
  const float* b_proj = (const float*)d_in[4];
  const float* W_gcn  = (const float*)d_in[5];
  const float* b_gcn  = (const float*)d_in[6];
  const float* W_dt   = (const float*)d_in[7];
  const float* b_dt   = (const float*)d_in[8];
  float* out = (float*)d_out;

  float* hp  = (float*)d_ws;                 // [8192,256]
  float* Yg  = hp + (size_t)NN * HIDD;       // [8192,256]
  float* fg  = Yg + (size_t)NN * HIDD;       // [8192,256]
  float* thr = fg + (size_t)NN * HIDD;       // [8192]

  // 1) hp = relu(h @ W_proj + b_proj)   (8192x256x256)
  {
    int tiles = (NN / 16) * (HIDD / 16);     // 8192 tiles
    gemm_bias_act<<<tiles / 8, 256, 0, stream>>>(
        h, W_proj, b_proj, hp, NN, HIDD, HIDD, 1, 1);
  }
  // 2) Y = hp @ W_gcn                   (no bias / no relu)
  {
    int tiles = (NN / 16) * (HIDD / 16);
    gemm_bias_act<<<tiles / 8, 256, 0, stream>>>(
        hp, W_gcn, nullptr, Yg, NN, HIDD, HIDD, 0, 0);
  }
  // 3) per-row top-500 threshold
  row_topk_thresh<<<NN, 256, 0, stream>>>(adj, simlar, thr);

  // 4) fused masked GCN GEMM: fg = relu(new_conn @ Y + b_gcn)
  {
    dim3 grid(NN / 128, HIDD / 64);          // 64 x 4
    gcn_gemm_fused<<<grid, 256, 0, stream>>>(adj, simlar, thr, Yg, b_gcn, fg);
  }
  // 5) out = relu(fg @ W_dt + b_dt)     (8192x256x64)
  {
    int tiles = (NN / 16) * (OUTD / 16);     // 2048 tiles
    gemm_bias_act<<<tiles / 8, 256, 0, stream>>>(
        fg, W_dt, b_dt, out, NN, HIDD, OUTD, 1, 1);
  }
}
